// LSTMEncoder_27462020891003
// MI455X (gfx1250) — compile-verified, matrix operand order checked by reading, not run on hardware
//
#include <hip/hip_runtime.h>
#include <hip/hip_bf16.h>

// ---------------- problem constants ----------------
#define BATCH 64
#define SEQT  1024
#define CDIM  512      // C == H
#define HDIM  512
#define G4    2048     // 4*H
#define TBROW 65536    // T*B
#define EPSLN 1e-5f

typedef __attribute__((ext_vector_type(16))) _Float16 v16h;
typedef __attribute__((ext_vector_type(8)))  _Float16 v8h;
typedef __attribute__((ext_vector_type(8)))  float    v8f;

union V16U { v16h v; v8h h[2]; };

// D = A*B + C, 16x16x32 f16 -> f32 accum (CDNA5 WMMA)
__device__ __forceinline__ v8f wmma32(v16h a, v16h b, v8f c) {
  return __builtin_amdgcn_wmma_f32_16x16x32_f16(false, a, false, b, (short)0, c,
                                                false, false);
}

// Load a 16xK=32 fragment (A or B operand) from a row-major [rows][ld] f16
// matrix at tile origin (r0,k0). CDNA5 layout: lane<16 -> row=lane, K 0..7 &
// 16..23 ; lane>=16 -> row=lane-16, K 8..15 & 24..31.
__device__ __forceinline__ v16h frag_rm(const _Float16* base, int ld, int r0, int k0) {
  int lane = threadIdx.x & 31;
  int half = lane >> 4;
  int r    = lane & 15;
  const _Float16* p = base + (size_t)(r0 + r) * ld + k0 + half * 8;
  V16U u;
  u.h[0] = *(const v8h*)p;
  u.h[1] = *(const v8h*)(p + 16);
  return u.v;
}

// Async global->LDS copy (CDNA5 TDM-adjacent data path, ASYNCcnt-tracked).
// Low 32 bits of a generic pointer to LDS are the LDS byte offset.
__device__ __forceinline__ void async_g2l_b128(void* lds, const void* gaddr) {
  unsigned lp = (unsigned)(size_t)lds;
  asm volatile("global_load_async_to_lds_b128 %0, %1, off"
               :: "v"(lp), "v"(gaddr) : "memory");
}
__device__ __forceinline__ void async_g2l_b32(void* lds, const void* gaddr) {
  unsigned lp = (unsigned)(size_t)lds;
  asm volatile("global_load_async_to_lds_b32 %0, %1, off"
               :: "v"(lp), "v"(gaddr) : "memory");
}
__device__ __forceinline__ void wait_async_le1() {
  asm volatile("s_wait_asynccnt 0x1" ::: "memory");
}
__device__ __forceinline__ void wait_async_0() {
  asm volatile("s_wait_asynccnt 0x0" ::: "memory");
}

__device__ __forceinline__ float sigf(float x) { return 1.0f / (1.0f + __expf(-x)); }

// ---------------- prep kernels ----------------
__global__ void k_cvt_f16(const float* __restrict__ s, _Float16* __restrict__ d, int n) {
  int i = blockIdx.x * 256 + threadIdx.x;
  if (i < n) d[i] = (_Float16)s[i];
}

__global__ void k_bias_fold(const float* __restrict__ a, const float* __restrict__ b,
                            float* __restrict__ o, int n) {
  int i = blockIdx.x * 256 + threadIdx.x;
  if (i < n) o[i] = a[i] + b[i];
}

// ---------------- LayerNorm(x) -> f16, transpose [B,T,C] -> [T,B,C] ----------------
__global__ __launch_bounds__(256) void k_ln_x(const float* __restrict__ x,
                                              const float* __restrict__ g,
                                              const float* __restrict__ b,
                                              _Float16* __restrict__ xn) {
  int row = blockIdx.x;            // row = bb*T + tt over [B,T]
  int bb = row >> 10;              // /SEQT
  int tt = row & (SEQT - 1);
  const float* xr = x + (size_t)row * CDIM;
  int tid = threadIdx.x;
  float v0 = xr[tid], v1 = xr[tid + 256];

  __shared__ float red1[8], red2[8];
  int lane = tid & 31, wv = tid >> 5;

  float s = v0 + v1;
  #pragma unroll
  for (int off = 16; off; off >>= 1) s += __shfl_xor(s, off, 32);
  if (lane == 0) red1[wv] = s;
  __syncthreads();
  float tot = 0.f;
  #pragma unroll
  for (int i = 0; i < 8; i++) tot += red1[i];
  float mean = tot * (1.0f / CDIM);

  float d0 = v0 - mean, d1 = v1 - mean;
  float vs = d0 * d0 + d1 * d1;
  #pragma unroll
  for (int off = 16; off; off >>= 1) vs += __shfl_xor(vs, off, 32);
  if (lane == 0) red2[wv] = vs;
  __syncthreads();
  float vtot = 0.f;
  #pragma unroll
  for (int i = 0; i < 8; i++) vtot += red2[i];
  float rs = rsqrtf(vtot * (1.0f / CDIM) + EPSLN);

  size_t o = ((size_t)tt * BATCH + bb) * CDIM;
  xn[o + tid]       = (_Float16)(d0 * rs * g[tid]       + b[tid]);
  xn[o + tid + 256] = (_Float16)(d1 * rs * g[tid + 256] + b[tid + 256]);
}

// ---------------- input projection: xg = xn @ Wih^T + (bih+bhh), both dirs ----------------
// A: xn [TBROW][512] f16 ; B: Wih [2048][512] f16 (row-major [N][K]) ; out f16.
// Block tile: 64(M) x 128(N), 8 waves, wave = 1 Mtile x 4 Ntiles, K chunked by 32.
// A tile staged via double-buffered async global->LDS copies.
__global__ __launch_bounds__(256) void k_xg_gemm(const _Float16* __restrict__ xn,
                                                 const _Float16* __restrict__ Wf,
                                                 const _Float16* __restrict__ Wb,
                                                 const float* __restrict__ biasf,
                                                 const float* __restrict__ biasb,
                                                 _Float16* __restrict__ xgf,
                                                 _Float16* __restrict__ xgb) {
  int bid = blockIdx.x;
  int dir = bid & 1;  bid >>= 1;
  int nt  = bid & 15;                 // N block (128 cols)
  int mt  = bid >> 4;                 // M block (64 rows)
  const _Float16* W   = dir ? Wb : Wf;
  const float*   bias = dir ? biasb : biasf;
  _Float16*      xg   = dir ? xgb : xgf;
  int m0 = mt * 64, n0 = nt * 128;

  __shared__ __align__(16) _Float16 As[2][64][32];

  int wv = threadIdx.x >> 5, lane = threadIdx.x & 31;
  int wm  = wv & 3;        // wave's M tile
  int grp = wv >> 2;       // wave's N group (0/1)

  v8f acc[4] = {};

  int lr  = threadIdx.x >> 2;        // 0..63
  int seg = (threadIdx.x & 3) * 8;   // 0,8,16,24 halves
  const _Float16* gbase = xn + (size_t)(m0 + lr) * CDIM + seg;

  // prologue: chunk 0 in flight
  async_g2l_b128(&As[0][lr][seg], gbase);

  for (int kc = 0; kc < 16; ++kc) {
    int cur = kc & 1;
    if (kc + 1 < 16) {
      async_g2l_b128(&As[cur ^ 1][lr][seg], gbase + (kc + 1) * 32);
      wait_async_le1();              // chunk kc landed (async loads retire in order)
    } else {
      wait_async_0();
    }
    __syncthreads();                 // all waves' chunk kc visible

    int k0 = kc * 32;
    int half = lane >> 4, r = lane & 15;
    const _Float16* p = &As[cur][16 * wm + r][half * 8];
    V16U ua; ua.h[0] = *(const v8h*)p; ua.h[1] = *(const v8h*)(p + 16);
    v16h bf[4];
    #pragma unroll
    for (int j = 0; j < 4; j++)
      bf[j] = frag_rm(W, CDIM, n0 + 16 * (4 * grp + j), k0);
    #pragma unroll
    for (int j = 0; j < 4; j++)
      acc[j] = wmma32(ua.v, bf[j], acc[j]);

    __syncthreads();                 // reads done before buffer is overwritten
  }

  int colL = lane & 15, halfL = lane >> 4;
  #pragma unroll
  for (int j = 0; j < 4; j++) {
    int n = n0 + 16 * (4 * grp + j) + colL;
    float bv = bias[n];
    #pragma unroll
    for (int v = 0; v < 8; v++) {
      int row = m0 + 16 * wm + v + 8 * halfL;
      xg[(size_t)row * G4 + n] = (_Float16)(acc[j][v] + bv);
    }
  }
}

// ---------------- recurrent scan (the sequential core) ----------------
// 8 workgroups: (dir, batch-tile-of-16). 32 waves; wave w owns h-columns
// 16w..16w+15 -> gate tiles at n = g*512 + 16w, so i/f/g/o line up in-register.
// h double-buffered in LDS; c lives in the WMMA C/D VGPR layout.
// Whh streams from L2 (2 MB f16/dir, resident in 192 MB L2); B frags are
// batched per K-chunk so the loads clause together ahead of 4 WMMAs.
__global__ __launch_bounds__(1024) void k_scan(const _Float16* __restrict__ xgf,
                                               const _Float16* __restrict__ xgb,
                                               const _Float16* __restrict__ Whf,
                                               const _Float16* __restrict__ Whb,
                                               _Float16* __restrict__ hsf,
                                               _Float16* __restrict__ hsb) {
  int dir = blockIdx.x >> 2;
  int m0  = (blockIdx.x & 3) * 16;                  // batch-row tile
  const _Float16* xg = dir ? xgb : xgf;
  const _Float16* Wh = dir ? Whb : Whf;
  _Float16*       hs = dir ? hsb : hsf;

  __shared__ __align__(16) _Float16 hbuf[2][16][HDIM];

  for (int i = threadIdx.x; i < 16 * HDIM; i += 1024)
    ((_Float16*)hbuf[0])[i] = (_Float16)0.0f;
  __syncthreads();

  int wv = threadIdx.x >> 5, lane = threadIdx.x & 31;
  int colL = lane & 15, halfL = lane >> 4;

  // warm L2 with this wave's Whh rows (global_prefetch_b8)
  #pragma unroll
  for (int g = 0; g < 4; g++)
    __builtin_prefetch(Wh + (size_t)(g * HDIM + 16 * wv) * HDIM, 0, 0);

  float cst[8];
  #pragma unroll
  for (int v = 0; v < 8; v++) cst[v] = 0.0f;

  for (int s = 0; s < SEQT; ++s) {
    int t = dir ? (SEQT - 1 - s) : s;
    int cur = s & 1, nxt = cur ^ 1;

    v8f acc[4] = {};
    for (int kc = 0; kc < 16; ++kc) {
      int k0 = kc * 32;
      const _Float16* ap = &hbuf[cur][colL][k0 + halfL * 8];
      V16U ua; ua.h[0] = *(const v8h*)ap; ua.h[1] = *(const v8h*)(ap + 16);
      v16h bf[4];
      #pragma unroll
      for (int g = 0; g < 4; g++)
        bf[g] = frag_rm(Wh, HDIM, g * HDIM + 16 * wv, k0);
      #pragma unroll
      for (int g = 0; g < 4; g++)
        acc[g] = wmma32(ua.v, bf[g], acc[g]);
    }

    size_t rowbase = (size_t)t * BATCH + m0;
    #pragma unroll
    for (int v = 0; v < 8; v++) {
      int row = v + 8 * halfL;
      int hc  = 16 * wv + colL;
      size_t gidx = (rowbase + row) * G4 + hc;
      float iv = sigf (acc[0][v] + (float)xg[gidx]);
      float fv = sigf (acc[1][v] + (float)xg[gidx + 512]);
      float gv = tanhf(acc[2][v] + (float)xg[gidx + 1024]);
      float ov = sigf (acc[3][v] + (float)xg[gidx + 1536]);
      float cn = fv * cst[v] + iv * gv;
      cst[v] = cn;
      _Float16 h16 = (_Float16)(ov * tanhf(cn));
      hbuf[nxt][row][hc] = h16;                       // next-step A operand
      hs[(rowbase + row) * HDIM + hc] = h16;          // full trajectory
    }
    __syncthreads();
  }
}

// ---------------- gate fusion + output LayerNorm ----------------
// sigma = sigmoid([fw|bw] @ Wg^T + bg); Hout = sigma*fw + (1-sigma)*bw; LN(Hout).
// One WG per 16 rows; 8 waves x 4 Ntiles cover N=512; K=1024 split fw/bw.
// A tile staged via double-buffered async global->LDS copies.
__global__ __launch_bounds__(256) void k_fuse(const _Float16* __restrict__ hsf,
                                              const _Float16* __restrict__ hsb,
                                              const _Float16* __restrict__ Wg16,
                                              const float* __restrict__ bg,
                                              const float* __restrict__ gH,
                                              const float* __restrict__ bH,
                                              float* __restrict__ out) {
  int r0 = blockIdx.x * 16;                           // rows r = t*B + b
  __shared__ __align__(16) _Float16 As[2][16][32];
  __shared__ float Hout[16][HDIM];

  int wv = threadIdx.x >> 5, lane = threadIdx.x & 31;
  int lrow = threadIdx.x >> 4;
  int lcol = (threadIdx.x & 15) * 2;
  v8f acc[4] = {};

  auto stage = [&](int kc, int buf) {
    int k0 = kc * 32;
    const _Float16* src = (k0 < 512) ? hsf : hsb;
    int kk = k0 & 511;
    async_g2l_b32(&As[buf][lrow][lcol],
                  src + (size_t)(r0 + lrow) * HDIM + kk + lcol);
  };

  stage(0, 0);
  for (int kc = 0; kc < 32; ++kc) {
    int cur = kc & 1;
    if (kc + 1 < 32) {
      stage(kc + 1, cur ^ 1);
      wait_async_le1();
    } else {
      wait_async_0();
    }
    __syncthreads();

    int k0 = kc * 32;
    int half = lane >> 4, r = lane & 15;
    const _Float16* p = &As[cur][r][half * 8];
    V16U ua; ua.h[0] = *(const v8h*)p; ua.h[1] = *(const v8h*)(p + 16);
    v16h bf[4];
    #pragma unroll
    for (int j = 0; j < 4; j++)
      bf[j] = frag_rm(Wg16, 2 * HDIM, 16 * (4 * wv + j), k0);
    #pragma unroll
    for (int j = 0; j < 4; j++)
      acc[j] = wmma32(ua.v, bf[j], acc[j]);

    __syncthreads();
  }

  int colL = lane & 15, halfL = lane >> 4;
  #pragma unroll
  for (int j = 0; j < 4; j++) {
    int n = 16 * (4 * wv + j) + colL;
    float bgv = bg[n];
    #pragma unroll
    for (int v = 0; v < 8; v++) {
      int row = v + 8 * halfL;
      size_t idx = (size_t)(r0 + row) * HDIM + n;
      float fw = (float)hsf[idx], bw = (float)hsb[idx];
      float sg = sigf(acc[j][v] + bgv);
      Hout[row][n] = sg * fw + (1.0f - sg) * bw;
    }
  }
  __syncthreads();

  // LayerNorm: 8 waves, 2 rows each, wave32 shfl reductions
  for (int rr = wv * 2; rr < wv * 2 + 2; ++rr) {
    float s = 0.f;
    #pragma unroll
    for (int i = 0; i < 16; i++) s += Hout[rr][lane * 16 + i];
    #pragma unroll
    for (int off = 16; off; off >>= 1) s += __shfl_xor(s, off, 32);
    float mean = s * (1.0f / HDIM);
    float vs = 0.f;
    #pragma unroll
    for (int i = 0; i < 16; i++) {
      float d = Hout[rr][lane * 16 + i] - mean;
      vs += d * d;
    }
    #pragma unroll
    for (int off = 16; off; off >>= 1) vs += __shfl_xor(vs, off, 32);
    float rs = rsqrtf(vs * (1.0f / HDIM) + EPSLN);

    int r = r0 + rr;
    int tt = r >> 6;          // / BATCH
    int bb = r & 63;
    float* op = out + ((size_t)bb * SEQT + tt) * HDIM;
    #pragma unroll
    for (int i = 0; i < 16; i++) {
      int col = lane * 16 + i;
      op[col] = (Hout[rr][col] - mean) * rs * gH[col] + bH[col];
    }
  }
}

// ---------------- host launcher ----------------
extern "C" void kernel_launch(void* const* d_in, const int* in_sizes, int n_in,
                              void* d_out, int out_size, void* d_ws, size_t ws_size,
                              hipStream_t stream) {
  (void)in_sizes; (void)n_in; (void)out_size; (void)ws_size;
  const float* x     = (const float*)d_in[0];
  // d_in[1] = ts (unused by forward)
  const float* Wih_f = (const float*)d_in[2];
  const float* Whh_f = (const float*)d_in[3];
  const float* bih_f = (const float*)d_in[4];
  const float* bhh_f = (const float*)d_in[5];
  const float* Wih_b = (const float*)d_in[6];
  const float* Whh_b = (const float*)d_in[7];
  const float* bih_b = (const float*)d_in[8];
  const float* bhh_b = (const float*)d_in[9];
  const float* Wg    = (const float*)d_in[10];
  const float* bg    = (const float*)d_in[11];
  const float* gx    = (const float*)d_in[12];
  const float* bx    = (const float*)d_in[13];
  const float* gH    = (const float*)d_in[14];
  const float* bH    = (const float*)d_in[15];
  float* out = (float*)d_out;

  char* w = (char*)d_ws;
  size_t off = 0;
  auto take = [&](size_t bytes) -> void* {
    void* p = w + off;
    off = (off + bytes + 255) & ~(size_t)255;
    return p;
  };

  _Float16* xn    = (_Float16*)take((size_t)TBROW * CDIM * 2);
  _Float16* Wf16  = (_Float16*)take((size_t)G4 * CDIM * 2);
  _Float16* Wb16  = (_Float16*)take((size_t)G4 * CDIM * 2);
  _Float16* Whf16 = (_Float16*)take((size_t)G4 * HDIM * 2);
  _Float16* Whb16 = (_Float16*)take((size_t)G4 * HDIM * 2);
  _Float16* Wg16  = (_Float16*)take((size_t)HDIM * 2 * HDIM * 2);
  float*    biasf = (float*)take((size_t)G4 * 4);
  float*    biasb = (float*)take((size_t)G4 * 4);
  _Float16* xgf   = (_Float16*)take((size_t)TBROW * G4 * 2);
  _Float16* xgb   = (_Float16*)take((size_t)TBROW * G4 * 2);
  _Float16* hsf   = (_Float16*)take((size_t)TBROW * HDIM * 2);
  _Float16* hsb   = (_Float16*)take((size_t)TBROW * HDIM * 2);

  const int nWih = G4 * CDIM;            // 1,048,576
  const int nWg  = HDIM * 2 * HDIM;      //   524,288
  k_cvt_f16<<<(nWih + 255) / 256, 256, 0, stream>>>(Wih_f, Wf16, nWih);
  k_cvt_f16<<<(nWih + 255) / 256, 256, 0, stream>>>(Wih_b, Wb16, nWih);
  k_cvt_f16<<<(nWih + 255) / 256, 256, 0, stream>>>(Whh_f, Whf16, nWih);
  k_cvt_f16<<<(nWih + 255) / 256, 256, 0, stream>>>(Whh_b, Whb16, nWih);
  k_cvt_f16<<<(nWg + 255) / 256, 256, 0, stream>>>(Wg, Wg16, nWg);
  k_bias_fold<<<(G4 + 255) / 256, 256, 0, stream>>>(bih_f, bhh_f, biasf, G4);
  k_bias_fold<<<(G4 + 255) / 256, 256, 0, stream>>>(bih_b, bhh_b, biasb, G4);

  k_ln_x<<<TBROW, 256, 0, stream>>>(x, gx, bx, xn);

  // 2 dirs * 1024 Mtiles * 16 Ntiles
  k_xg_gemm<<<2 * 1024 * 16, 256, 0, stream>>>(xn, Wf16, Wb16, biasf, biasb, xgf, xgb);

  // 2 dirs * 4 batch tiles, 32 waves each
  k_scan<<<8, 1024, 0, stream>>>(xgf, xgb, Whf16, Whb16, hsf, hsb);

  // 65536/16 row blocks
  k_fuse<<<TBROW / 16, 256, 0, stream>>>(hsf, hsb, Wg16, bg, gH, bH, out);
}
